// ComENet_76347338654246
// MI455X (gfx1250) — matrix operand order repaired and checked by reading, not compiled, not run on hardware
//
#include <hip/hip_runtime.h>

// ---------------------------------------------------------------------------
// ComENet forward for MI455X (gfx1250, wave32, WMMA).
// Dense matmuls: bf16 WMMA (v_wmma_f32_16x16x32_bf16), fp32 accumulate.
// Graph ops: packed-u64 atomicMin argmin (deterministic), f32 atomic scatter.
// ---------------------------------------------------------------------------

#define HDIM 256
#define MDIM 64
#define NLAYERS 4
#define NOUT 3
#define PI_F 3.14159265358979323846f

typedef __bf16 bf16_t;
typedef bf16_t v16bf __attribute__((ext_vector_type(16)));
typedef float  v8f   __attribute__((ext_vector_type(8)));

__device__ __forceinline__ unsigned short f2bf(float f) {
  unsigned u = __float_as_uint(f);
  u += 0x7fffu + ((u >> 16) & 1u);           // round-to-nearest-even
  return (unsigned short)(u >> 16);
}
__device__ __forceinline__ float siluf(float v) { return v / (1.f + expf(-v)); }

// --------------------------- utility kernels -------------------------------

__global__ void k_fill_u64(unsigned long long* p, unsigned long long v, long long n) {
  long long i = (long long)blockIdx.x * blockDim.x + threadIdx.x;
  if (i < n) p[i] = v;
}
__global__ void k_fill_f32(float* p, float v, long long n) {
  long long i = (long long)blockIdx.x * blockDim.x + threadIdx.x;
  if (i < n) p[i] = v;
}
__global__ void k_fill_i32(int* p, int v, long long n) {
  long long i = (long long)blockIdx.x * blockDim.x + threadIdx.x;
  if (i < n) p[i] = v;
}
__global__ void k_cvt_bf16(unsigned short* __restrict__ d, const float* __restrict__ s, long long n) {
  long long i = (long long)blockIdx.x * blockDim.x + threadIdx.x;
  if (i < n) d[i] = f2bf(s[i]);
}

// --------------------------- weight packing --------------------------------
// Pack row-major fp32 W[K][Nc] into the WMMA B-fragment layout:
// out[((kk*ntiles + bn)*32 + lane)*16 + e], with N = bn*16 + lane%16,
// K = kk*32 + 8*(lane/16) + e + 8*(e/8).  K zero-padded up to a multiple of 32.
__global__ void k_pack_b(const float* __restrict__ W, unsigned short* __restrict__ out,
                         int K, int Nc) {
  int t = blockIdx.x * blockDim.x + threadIdx.x;
  int ktiles = (K + 31) >> 5;
  int ntiles = Nc >> 4;
  if (t >= ktiles * ntiles * 32) return;
  int lane = t & 31;
  int tile = t >> 5;
  int bn = tile % ntiles;
  int kk = tile / ntiles;
  int col = bn * 16 + (lane & 15);
  int kbase = kk * 32 + ((lane >> 4) << 3);
  unsigned short* o = out + (size_t)t * 16;
#pragma unroll
  for (int e = 0; e < 16; ++e) {
    int k = kbase + e + ((e >> 3) << 3);
    float v = (k < K) ? W[(size_t)k * Nc + col] : 0.f;
    o[e] = f2bf(v);
  }
}

// ------------------------------ WMMA GEMM ----------------------------------
// C[m, col0+n] = sum_k A[m,k]*B[k,n] + bias[n].  A: bf16 row-major [rows,K]
// (K multiple of 32).  B: packed by k_pack_b.  One 16x16 tile per wave.
__global__ void __launch_bounds__(256)
k_gemm_bf16(const unsigned short* __restrict__ A, const unsigned short* __restrict__ Bp,
            const float* __restrict__ bias, float* __restrict__ C,
            int rows, int K, int Ncols, int ldc, int col0) {
  const int wave = threadIdx.x >> 5;
  const int lane = threadIdx.x & 31;
  const int hi   = lane >> 4;                 // half-wave select
  const int mt   = blockIdx.x * 8 + wave;
  const int bn   = blockIdx.y;
  const int r0   = mt << 4;
  if (r0 >= rows) return;
  int arow = r0 + (lane & 15);
  if (arow >= rows) arow = rows - 1;          // clamp; masked on store

  const unsigned short* aptr = A + (size_t)arow * K + (hi << 3);
  const int ntiles = Ncols >> 4;
  const unsigned short* bptr = Bp + ((size_t)bn * 32 + lane) * 16;
  const size_t bstep = (size_t)ntiles * 32 * 16;

  v8f acc = {};
  const int ksteps = K >> 5;
  for (int kk = 0; kk < ksteps; ++kk) {
    v16bf a, b;
    // A fragment: lane holds row M=lane%16; K = kstep*32 + {0..7,16..23} + 8*hi
    *(uint4*)&a       = *(const uint4*)(aptr);
    *((uint4*)&a + 1) = *(const uint4*)(aptr + 16);
    // B fragment: 16 contiguous packed bf16 per lane
    *(uint4*)&b       = *(const uint4*)(bptr);
    *((uint4*)&b + 1) = *(const uint4*)(bptr + 8);
    __builtin_prefetch(aptr + 32, 0, 1);      // global_prefetch_b8
    acc = __builtin_amdgcn_wmma_f32_16x16x32_bf16(
        false, a, false, b, (short)0, acc, false, false);
    aptr += 32;
    bptr += bstep;
  }

  const int nloc = lane & 15;
  const float bv = bias ? bias[bn * 16 + nloc] : 0.f;
  const int col = col0 + bn * 16 + nloc;
  const int mbase = r0 + (hi << 3);           // D: VGPR r -> M = r + 8*hi
#pragma unroll
  for (int r = 0; r < 8; ++r) {
    int m = mbase + r;
    if (m < rows) C[(size_t)m * ldc + col] = acc[r] + bv;
  }
}

// ------------------------------ geometry -----------------------------------

struct F3 { float x, y, z; };
__device__ __forceinline__ F3 cross3(F3 a, F3 b) {
  return {a.y * b.z - a.z * b.y, a.z * b.x - a.x * b.z, a.x * b.y - a.y * b.x};
}
__device__ __forceinline__ float dot3(F3 a, F3 b) { return a.x * b.x + a.y * b.y + a.z * b.z; }

__device__ __forceinline__ unsigned long long packdm(float d, int e) {
  return ((unsigned long long)__float_as_uint(d) << 32) | (unsigned)e;
}

// pass 1: edge vectors, distances, argmin0 (i-keyed and j-keyed)
__global__ void k_geom1(const float* __restrict__ pos, const int* __restrict__ ei, int E,
                        float* vx, float* vy, float* vz, float* dist,
                        unsigned long long* p0i, unsigned long long* p0j) {
  int e = blockIdx.x * blockDim.x + threadIdx.x;
  if (e >= E) return;
  int jn = ei[e], in = ei[E + e];
  float dx = pos[jn * 3 + 0] - pos[in * 3 + 0];
  float dy = pos[jn * 3 + 1] - pos[in * 3 + 1];
  float dz = pos[jn * 3 + 2] - pos[in * 3 + 2];
  float d = sqrtf(dx * dx + dy * dy + dz * dz);
  vx[e] = dx; vy[e] = dy; vz[e] = dz; dist[e] = d;
  unsigned long long pk = packdm(d, e);
  atomicMin(&p0i[in], pk);
  atomicMin(&p0j[jn], pk);
}

__global__ void k_extract0(const unsigned long long* p0i, const unsigned long long* p0j,
                           const int* __restrict__ ei, int E, int N,
                           int* a0i, int* a0j, int* n0i, int* n0j,
                           int* flag0i, int* flag0j) {
  int n = blockIdx.x * blockDim.x + threadIdx.x;
  if (n >= N) return;
  unsigned long long v = p0i[n];
  int a = (v == 0xFFFFFFFFFFFFFFFFull) ? 0 : (int)(v & 0xFFFFFFFFu);
  a0i[n] = a; n0i[n] = ei[a];            // n0 = j[argmin0]
  flag0i[a] = 1;                          // matches .at[argmin0].set(CUTOFF)
  v = p0j[n];
  a = (v == 0xFFFFFFFFFFFFFFFFull) ? 0 : (int)(v & 0xFFFFFFFFu);
  a0j[n] = a; n0j[n] = ei[E + a];        // n0_j = i[argmin0_j]
  flag0j[a] = 1;
}

// pass 2: argmin1 with +CUTOFF penalty on previous winners
__global__ void k_geom2(const int* __restrict__ ei, int E, const float* __restrict__ dist,
                        const int* flag0i, const int* flag0j,
                        unsigned long long* p1i, unsigned long long* p1j) {
  int e = blockIdx.x * blockDim.x + threadIdx.x;
  if (e >= E) return;
  int jn = ei[e], in = ei[E + e];
  float d = dist[e];
  atomicMin(&p1i[in], packdm(d + (flag0i[e] ? 8.0f : 0.f), e));
  atomicMin(&p1j[jn], packdm(d + (flag0j[e] ? 8.0f : 0.f), e));
}

__global__ void k_extract1(const unsigned long long* p1i, const unsigned long long* p1j,
                           int N, int* a1i, int* a1j) {
  int n = blockIdx.x * blockDim.x + threadIdx.x;
  if (n >= N) return;
  unsigned long long v = p1i[n];
  a1i[n] = (v == 0xFFFFFFFFFFFFFFFFull) ? 0 : (int)(v & 0xFFFFFFFFu);
  v = p1j[n];
  a1j[n] = (v == 0xFFFFFFFFFFFFFFFFull) ? 0 : (int)(v & 0xFFFFFFFFu);
}

// per-edge angular + radial basis -> bf16 padded feature rows (12->32, 6->32)
__global__ void k_features(const int* __restrict__ ei, int E,
                           const float* vx, const float* vy, const float* vz,
                           const float* dist,
                           const int* a0i, const int* a1i, const int* a0j, const int* a1j,
                           const int* n0i, const int* n0j,
                           unsigned short* __restrict__ f1p, unsigned short* __restrict__ f2p) {
  int e = blockIdx.x * blockDim.x + threadIdx.x;
  if (e >= E) return;
  int jn = ei[e], in = ei[E + e];
  F3 ji = {vx[e], vy[e], vz[e]};
  float d = dist[e];

  int e_n0 = a0i[in], e_n1 = a1i[in];
  int idx_iref = (n0i[in] == jn) ? e_n1 : e_n0;
  int idx_jref = (n0j[jn] == in) ? a1j[jn] : a0j[jn];

  F3 in0  = {vx[e_n0], vy[e_n0], vz[e_n0]};
  F3 in1  = {vx[e_n1], vy[e_n1], vz[e_n1]};
  F3 iref = {vx[idx_iref], vy[idx_iref], vz[idx_iref]};
  F3 jref = {vx[idx_jref], vy[idx_jref], vz[idx_jref]};
  F3 nji = {-ji.x, -ji.y, -ji.z};

  // theta
  float av = dot3(nji, in0);
  F3 cr = cross3(nji, in0);
  float bv = sqrtf(dot3(cr, cr));
  float theta = atan2f(bv, av); if (theta < 0.f) theta += PI_F;
  // phi
  F3 p1 = cross3(nji, in0), p2 = cross3(nji, in1);
  av = dot3(p1, p2); bv = dot3(cross3(p1, p2), ji) / d;
  float phi = atan2f(bv, av); if (phi < 0.f) phi += PI_F;
  // tau
  p1 = cross3(ji, jref); p2 = cross3(ji, iref);
  av = dot3(p1, p2); bv = dot3(cross3(p1, p2), ji) / d;
  float tau = atan2f(bv, av); if (tau < 0.f) tau += PI_F;

  // radial basis (l=0,1; n=1..3)
  const float PIN[3] = {3.14159265f, 6.28318531f, 9.42477796f};
  const float Z1c[3] = {4.4934095f, 7.7252518f, 10.904122f};
  const float N1c[3] = {6.5101233f, 11.016225f, 15.485661f};
  float dd = d * 0.125f;
  float r0[3], r1[3];
#pragma unroll
  for (int n = 0; n < 3; ++n) {
    r0[n] = 1.41421356f * sinf(dd * PIN[n]) / dd;
    float u = dd * Z1c[n];
    r1[n] = N1c[n] * (sinf(u) / (u * u) - cosf(u) / u);
  }
  float st = sinf(theta);
  float sbf[4] = {0.28209479f, 0.48860252f * st * sinf(phi),
                  0.48860252f * cosf(theta), 0.48860252f * st * cosf(phi)};
  float cbf[2] = {0.28209479f, 0.48860252f * cosf(tau)};

  unsigned short* o1 = f1p + (size_t)e * 32;
  unsigned short* o2 = f2p + (size_t)e * 32;
#pragma unroll
  for (int l = 0; l < 2; ++l)
#pragma unroll
    for (int n = 0; n < 3; ++n) {
      float r = l ? r1[n] : r0[n];
      o1[l * 6 + n * 2 + 0] = f2bf(r * sbf[l * 2 + 0]);
      o1[l * 6 + n * 2 + 1] = f2bf(r * sbf[l * 2 + 1]);
      o2[l * 3 + n] = f2bf(r * cbf[l]);
    }
#pragma unroll
  for (int k = 12; k < 32; ++k) o1[k] = 0;
#pragma unroll
  for (int k = 6; k < 32; ++k) o2[k] = 0;
}

// ------------------------------ node kernels -------------------------------

__global__ void k_embed(const float* __restrict__ emb, const int* __restrict__ z,
                        float* __restrict__ x) {
  int n = blockIdx.x, c = threadIdx.x;
  float v = emb[(size_t)z[n] * HDIM + c];
  x[(size_t)n * HDIM + c] = siluf(v);
}

// one block per graph (A contiguous atoms), thread = channel
__global__ void k_graphnorm_silu(const float* __restrict__ x,
                                 const float* __restrict__ w, const float* __restrict__ b,
                                 const float* __restrict__ ms,
                                 float* __restrict__ xn, unsigned short* __restrict__ xnbf,
                                 int A) {
  int g = blockIdx.x, c = threadIdx.x;
  const float* xg = x + (size_t)g * A * HDIM;
  float s = 0.f;
  for (int a = 0; a < A; ++a) s += xg[(size_t)a * HDIM + c];
  float sub = (s / A) * ms[c];
  float var = 0.f;
  for (int a = 0; a < A; ++a) { float o = xg[(size_t)a * HDIM + c] - sub; var += o * o; }
  var /= A;
  float scale = w[c] * rsqrtf(var + 1e-5f);
  float bb = b[c];
  float* xng = xn + (size_t)g * A * HDIM;
  unsigned short* xnbg = xnbf + (size_t)g * A * HDIM;
  for (int a = 0; a < A; ++a) {
    float v = (xg[(size_t)a * HDIM + c] - sub) * scale + bb;
    float sv = siluf(v);
    xng[(size_t)a * HDIM + c] = sv;
    xnbg[(size_t)a * HDIM + c] = f2bf(sv);
  }
}

// fused: f = t[e,0:64] @ w2[64,256]; msg[i] += xn[j] * f  (block per edge)
__global__ void __launch_bounds__(256)
k_edge_messages(const int* __restrict__ ei, int E,
                const float* __restrict__ t1, const float* __restrict__ t2,
                const float* __restrict__ w2a, const float* __restrict__ w2b,
                const float* __restrict__ xn,
                float* __restrict__ msg1, float* __restrict__ msg2) {
  __shared__ float s1[MDIM], s2[MDIM];
  int e = blockIdx.x, c = threadIdx.x;
  if (c < MDIM) {
    s1[c] = t1[(size_t)e * MDIM + c];
    s2[c] = t2[(size_t)e * MDIM + c];
  }
  __syncthreads();
  int jn = ei[e], in = ei[E + e];
  float f1 = 0.f, f2 = 0.f;
#pragma unroll 8
  for (int m = 0; m < MDIM; ++m) {
    f1 += s1[m] * w2a[m * HDIM + c];
    f2 += s2[m] * w2b[m * HDIM + c];
  }
  float xv = xn[(size_t)jn * HDIM + c];
  atomicAdd(&msg1[(size_t)in * HDIM + c], xv * f1);
  atomicAdd(&msg2[(size_t)in * HDIM + c], xv * f2);
}

__global__ void k_add_bf(float* __restrict__ h, const float* __restrict__ xn,
                         unsigned short* __restrict__ hbf, long long n) {
  long long i = (long long)blockIdx.x * blockDim.x + threadIdx.x;
  if (i >= n) return;
  float v = h[i] + xn[i];
  h[i] = v; hbf[i] = f2bf(v);
}
__global__ void k_silu_add_bf(float* __restrict__ h, const float* __restrict__ g,
                              unsigned short* __restrict__ hbf, long long n) {
  long long i = (long long)blockIdx.x * blockDim.x + threadIdx.x;
  if (i >= n) return;
  float v = siluf(g[i]) + h[i];
  h[i] = v; hbf[i] = f2bf(v);
}
__global__ void k_silu_store(float* __restrict__ x, const float* __restrict__ g, long long n) {
  long long i = (long long)blockIdx.x * blockDim.x + threadIdx.x;
  if (i < n) x[i] = siluf(g[i]);
}

__global__ void k_linout(const float* __restrict__ x, const float* __restrict__ wv,
                         const float* __restrict__ bv, float* __restrict__ nodeval) {
  __shared__ float red[HDIM];
  int n = blockIdx.x, c = threadIdx.x;
  red[c] = x[(size_t)n * HDIM + c] * wv[c];
  __syncthreads();
  for (int s = HDIM / 2; s > 0; s >>= 1) {
    if (c < s) red[c] += red[c + s];
    __syncthreads();
  }
  if (c == 0) nodeval[n] = red[0] + bv[0];
}

__global__ void k_graphsum(const float* __restrict__ nodeval, float* __restrict__ out, int A) {
  int g = blockIdx.x;
  if (threadIdx.x == 0) {
    float s = 0.f;
    for (int a = 0; a < A; ++a) s += nodeval[(size_t)g * A + a];
    out[g] = s;
  }
}

// ------------------------------ host glue ----------------------------------

extern "C" void kernel_launch(void* const* d_in, const int* in_sizes, int n_in,
                              void* d_out, int out_size, void* d_ws, size_t ws_size,
                              hipStream_t stream) {
  (void)n_in; (void)ws_size;
  const float* pos       = (const float*)d_in[0];
  const float* emb       = (const float*)d_in[1];
  const float* conv1_w1  = (const float*)d_in[2];
  const float* conv1_w2  = (const float*)d_in[3];
  const float* conv2_w1  = (const float*)d_in[4];
  const float* conv2_w2  = (const float*)d_in[5];
  const float* lin1_w    = (const float*)d_in[6];
  const float* lin1_b    = (const float*)d_in[7];
  const float* lin2_w    = (const float*)d_in[8];
  const float* lin2_b    = (const float*)d_in[9];
  const float* lincat_w  = (const float*)d_in[10];
  const float* lincat_b  = (const float*)d_in[11];
  const float* norm_w    = (const float*)d_in[12];
  const float* norm_b    = (const float*)d_in[13];
  const float* norm_ms   = (const float*)d_in[14];
  const float* blk_w     = (const float*)d_in[15];
  const float* blk_b     = (const float*)d_in[16];
  const float* final_w   = (const float*)d_in[17];
  const float* final_b   = (const float*)d_in[18];
  const float* out_w     = (const float*)d_in[19];
  const float* out_b     = (const float*)d_in[20];
  const float* lout_w    = (const float*)d_in[21];
  const float* lout_b    = (const float*)d_in[22];
  const int*   z         = (const int*)d_in[23];
  const int*   ei        = (const int*)d_in[25];

  const int N = in_sizes[23];
  const int E = in_sizes[25] / 2;
  const int G = out_size;               // [G,1] energies
  const int A = N / G;                  // atoms per graph (contiguous batch)
  float* out = (float*)d_out;

  // ---- workspace carve-up (256B aligned) ----
  char* base = (char*)d_ws;
  size_t off = 0;
  auto alloc = [&](size_t bytes) -> void* {
    size_t o = (off + 255) & ~(size_t)255;
    off = o + bytes;
    return (void*)(base + o);
  };
  const size_t NE = (size_t)E, NN = (size_t)N;
  float* vx   = (float*)alloc(NE * 4);
  float* vy   = (float*)alloc(NE * 4);
  float* vz   = (float*)alloc(NE * 4);
  float* dist = (float*)alloc(NE * 4);
  unsigned long long* pmin = (unsigned long long*)alloc(4 * NN * 8);
  unsigned long long *p0i = pmin, *p1i = pmin + NN, *p0j = pmin + 2 * NN, *p1j = pmin + 3 * NN;
  int* ibuf = (int*)alloc(6 * NN * 4);
  int *a0i = ibuf, *a1i = ibuf + NN, *a0j = ibuf + 2 * NN, *a1j = ibuf + 3 * NN,
      *n0i = ibuf + 4 * NN, *n0j = ibuf + 5 * NN;
  int* flags = (int*)alloc(2 * NE * 4);
  int *flag0i = flags, *flag0j = flags + NE;
  unsigned short* feat1p = (unsigned short*)alloc(NE * 32 * 2);
  unsigned short* feat2p = (unsigned short*)alloc(NE * 32 * 2);
  float* t1 = (float*)alloc(NE * MDIM * 4);
  float* t2 = (float*)alloc(NE * MDIM * 4);
  float* x    = (float*)alloc(NN * HDIM * 4);
  float* xn   = (float*)alloc(NN * HDIM * 4);
  unsigned short* xnbf = (unsigned short*)alloc(NN * HDIM * 2);
  float* msg  = (float*)alloc(2 * NN * HDIM * 4);      // msg1 | msg2
  float *msg1 = msg, *msg2 = msg + NN * HDIM;
  unsigned short* msgbf = (unsigned short*)alloc(2 * NN * HDIM * 2);
  unsigned short *msg1bf = msgbf, *msg2bf = msgbf + NN * HDIM;
  float* hcat = (float*)alloc(NN * 2 * HDIM * 4);
  unsigned short* hcatbf = (unsigned short*)alloc(NN * 2 * HDIM * 2);
  float* h    = (float*)alloc(NN * HDIM * 4);
  unsigned short* hbf = (unsigned short*)alloc(NN * HDIM * 2);
  float* gbuf = (float*)alloc(NN * HDIM * 4);
  unsigned short* xbf = (unsigned short*)alloc(NN * HDIM * 2);
  float* nodeval = (float*)alloc(NN * 4);
  // packed weights (bf16, WMMA B layout)
  unsigned short* c1w1p  = (unsigned short*)alloc((size_t)NLAYERS * 32 * MDIM * 2);
  unsigned short* c2w1p  = (unsigned short*)alloc((size_t)NLAYERS * 32 * MDIM * 2);
  unsigned short* lin1p  = (unsigned short*)alloc((size_t)NLAYERS * HDIM * HDIM * 2);
  unsigned short* lin2p  = (unsigned short*)alloc((size_t)NLAYERS * HDIM * HDIM * 2);
  unsigned short* lincatp= (unsigned short*)alloc((size_t)NLAYERS * 2 * HDIM * HDIM * 2);
  unsigned short* blkp   = (unsigned short*)alloc((size_t)NLAYERS * NOUT * HDIM * HDIM * 2);
  unsigned short* finalp = (unsigned short*)alloc((size_t)NLAYERS * HDIM * HDIM * 2);
  unsigned short* outp   = (unsigned short*)alloc((size_t)NOUT * HDIM * HDIM * 2);

  auto cdiv = [](long long a, long long b) { return (unsigned)((a + b - 1) / b); };
  auto pack = [&](const float* W, unsigned short* dst, int K, int Nc) {
    int total = ((K + 31) / 32) * (Nc / 16) * 32;
    k_pack_b<<<cdiv(total, 256), 256, 0, stream>>>(W, dst, K, Nc);
  };
  auto gemm = [&](const unsigned short* Ab, const unsigned short* Bb, const float* bias,
                  float* Cb, int rows, int Kd, int Nc, int ldc, int col0) {
    dim3 g(cdiv(rows, 128), (unsigned)(Nc / 16));
    k_gemm_bf16<<<g, 256, 0, stream>>>(Ab, Bb, bias, Cb, rows, Kd, Nc, ldc, col0);
  };
  auto cvt = [&](unsigned short* dst, const float* src, long long n) {
    k_cvt_bf16<<<cdiv(n, 256), 256, 0, stream>>>(dst, src, n);
  };

  // ---- pack all weights (cheap, deterministic, every call) ----
  for (int l = 0; l < NLAYERS; ++l) {
    pack(conv1_w1 + (size_t)l * 12 * MDIM, c1w1p + (size_t)l * 32 * MDIM, 12, MDIM);
    pack(conv2_w1 + (size_t)l * 6  * MDIM, c2w1p + (size_t)l * 32 * MDIM, 6, MDIM);
    pack(lin1_w   + (size_t)l * HDIM * HDIM, lin1p + (size_t)l * HDIM * HDIM, HDIM, HDIM);
    pack(lin2_w   + (size_t)l * HDIM * HDIM, lin2p + (size_t)l * HDIM * HDIM, HDIM, HDIM);
    pack(lincat_w + (size_t)l * 2 * HDIM * HDIM, lincatp + (size_t)l * 2 * HDIM * HDIM, 2 * HDIM, HDIM);
    for (int m = 0; m < NOUT; ++m)
      pack(blk_w + ((size_t)l * NOUT + m) * HDIM * HDIM,
           blkp  + ((size_t)l * NOUT + m) * HDIM * HDIM, HDIM, HDIM);
    pack(final_w + (size_t)l * HDIM * HDIM, finalp + (size_t)l * HDIM * HDIM, HDIM, HDIM);
  }
  for (int m = 0; m < NOUT; ++m)
    pack(out_w + (size_t)m * HDIM * HDIM, outp + (size_t)m * HDIM * HDIM, HDIM, HDIM);

  // ---- geometry / neighbor selection ----
  k_fill_u64<<<cdiv(4LL * N, 256), 256, 0, stream>>>(pmin, 0xFFFFFFFFFFFFFFFFull, 4LL * N);
  k_fill_i32<<<cdiv(2LL * E, 256), 256, 0, stream>>>(flags, 0, 2LL * E);
  k_geom1<<<cdiv(E, 256), 256, 0, stream>>>(pos, ei, E, vx, vy, vz, dist, p0i, p0j);
  k_extract0<<<cdiv(N, 256), 256, 0, stream>>>(p0i, p0j, ei, E, N, a0i, a0j, n0i, n0j, flag0i, flag0j);
  k_geom2<<<cdiv(E, 256), 256, 0, stream>>>(ei, E, dist, flag0i, flag0j, p1i, p1j);
  k_extract1<<<cdiv(N, 256), 256, 0, stream>>>(p1i, p1j, N, a1i, a1j);
  k_features<<<cdiv(E, 256), 256, 0, stream>>>(ei, E, vx, vy, vz, dist,
                                               a0i, a1i, a0j, a1j, n0i, n0j, feat1p, feat2p);

  // ---- initial embedding ----
  k_embed<<<N, HDIM, 0, stream>>>(emb, z, x);

  const long long NH = (long long)N * HDIM;
  // ---- interaction layers ----
  for (int l = 0; l < NLAYERS; ++l) {
    k_graphnorm_silu<<<G, HDIM, 0, stream>>>(x, norm_w + l * HDIM, norm_b + l * HDIM,
                                             norm_ms + l * HDIM, xn, xnbf, A);
    // edge basis projections (WMMA, K padded to 32)
    gemm(feat1p, c1w1p + (size_t)l * 32 * MDIM, nullptr, t1, E, 32, MDIM, MDIM, 0);
    gemm(feat2p, c2w1p + (size_t)l * 32 * MDIM, nullptr, t2, E, 32, MDIM, MDIM, 0);
    // scatter messages
    k_fill_f32<<<cdiv(2 * NH, 256), 256, 0, stream>>>(msg, 0.f, 2 * NH);
    k_edge_messages<<<E, HDIM, 0, stream>>>(ei, E, t1, t2,
                                            conv1_w2 + (size_t)l * MDIM * HDIM,
                                            conv2_w2 + (size_t)l * MDIM * HDIM,
                                            xn, msg1, msg2);
    cvt(msgbf, msg, 2 * NH);
    // h1 | h2 -> hcat, then lincat
    gemm(msg1bf, lin1p + (size_t)l * HDIM * HDIM, lin1_b + l * HDIM, hcat, N, HDIM, HDIM, 2 * HDIM, 0);
    gemm(msg2bf, lin2p + (size_t)l * HDIM * HDIM, lin2_b + l * HDIM, hcat, N, HDIM, HDIM, 2 * HDIM, HDIM);
    cvt(hcatbf, hcat, 2 * NH);
    gemm(hcatbf, lincatp + (size_t)l * 2 * HDIM * HDIM, lincat_b + l * HDIM, h, N, 2 * HDIM, HDIM, HDIM, 0);
    k_add_bf<<<cdiv(NH, 256), 256, 0, stream>>>(h, xn, hbf, NH);
    // residual MLP blocks
    for (int m = 0; m < NOUT; ++m) {
      gemm(hbf, blkp + ((size_t)l * NOUT + m) * HDIM * HDIM,
           blk_b + ((size_t)l * NOUT + m) * HDIM, gbuf, N, HDIM, HDIM, HDIM, 0);
      k_silu_add_bf<<<cdiv(NH, 256), 256, 0, stream>>>(h, gbuf, hbf, NH);
    }
    gemm(hbf, finalp + (size_t)l * HDIM * HDIM, final_b + l * HDIM, x, N, HDIM, HDIM, HDIM, 0);
  }

  // ---- output head ----
  for (int m = 0; m < NOUT; ++m) {
    cvt(xbf, x, NH);
    gemm(xbf, outp + (size_t)m * HDIM * HDIM, out_b + m * HDIM, gbuf, N, HDIM, HDIM, HDIM, 0);
    k_silu_store<<<cdiv(NH, 256), 256, 0, stream>>>(x, gbuf, NH);
  }
  k_linout<<<N, HDIM, 0, stream>>>(x, lout_w, lout_b, nodeval);
  k_graphsum<<<G, 32, 0, stream>>>(nodeval, out, A);
}
